// attention_block_53386443489782
// MI455X (gfx1250) — compile-verified, matrix operand order checked by reading, not run on hardware
//
#include <hip/hip_runtime.h>
#include <hip/hip_bf16.h>
#include <math.h>

#define H_DIM 1024
#define B_DIM 8
#define S_DIM 2048
#define EPSV  1e-5f

typedef __bf16        v16bf __attribute__((ext_vector_type(16)));
typedef float         v8f   __attribute__((ext_vector_type(8)));
typedef unsigned int  u32x4 __attribute__((ext_vector_type(4)));

// ---------- helpers ----------
static __device__ __forceinline__ unsigned short f2bf(float f) {
  unsigned u = __float_as_uint(f);
  u += 0x7FFFu + ((u >> 16) & 1u);          // round-to-nearest-even
  return (unsigned short)(u >> 16);
}
static __device__ __forceinline__ unsigned pack2bf(float a, float b) {
  return (unsigned)f2bf(a) | ((unsigned)f2bf(b) << 16);
}
static __device__ __forceinline__ float sigmoidf_(float x) {
  return 1.0f / (1.0f + __expf(-x));
}
// order-preserving uint encoding of float (for deterministic atomicMax)
static __device__ __forceinline__ unsigned f32_ord(float f) {
  unsigned u = __float_as_uint(f);
  return (u & 0x80000000u) ? ~u : (u | 0x80000000u);
}
static __device__ __forceinline__ float ord_f32(unsigned u) {
  u = (u & 0x80000000u) ? (u & 0x7FFFFFFFu) : ~u;
  return __uint_as_float(u);
}

union FragU { u32x4 q[2]; v16bf v; };

// Load one 16x32 (A) or 32x16-as-[N,K] (B) bf16 WMMA fragment.
// Lane l<16: row (M or N) = l, K = 0..15 ; lane l>=16: row = l-16, K = 16..31.
// => per lane: 32 contiguous bytes at (row*ld + khalf*16).
static __device__ __forceinline__ v16bf load_frag(const unsigned short* tile, int ld) {
  const int lane = threadIdx.x & 31;
  const unsigned short* p = tile + (size_t)(lane & 15) * ld + ((lane >> 4) << 4);
  FragU u;
  u.q[0] = *(const u32x4*)p;
  u.q[1] = *(const u32x4*)(p + 8);
  return u.v;
}

#define WMMA_BF16(a, b, c) \
  __builtin_amdgcn_wmma_f32_16x16x32_bf16(false, (a), false, (b), (short)0, (c), false, false)

// Shared inner loop: wave computes a 64(M) x 64(N) tile, lda == ldb == K.
// 16 WMMAs per k-step vs 16 b128 loads -> 1.0 WMMA per load.
static __device__ __forceinline__ void gemm_tile_loop(
    const unsigned short* __restrict__ A0,
    const unsigned short* __restrict__ B0, int K, v8f acc[4][4]) {
  for (int k = 0; k < K; k += 32) {
    __builtin_prefetch(A0 + k + 128, 0, 3);
    __builtin_prefetch(B0 + k + 128, 0, 3);
    v16bf a0 = load_frag(A0 + k, K);
    v16bf a1 = load_frag(A0 + (size_t)16 * K + k, K);
    v16bf a2 = load_frag(A0 + (size_t)32 * K + k, K);
    v16bf a3 = load_frag(A0 + (size_t)48 * K + k, K);
    v16bf b0 = load_frag(B0 + k, K);
    v16bf b1 = load_frag(B0 + (size_t)16 * K + k, K);
    v16bf b2 = load_frag(B0 + (size_t)32 * K + k, K);
    v16bf b3 = load_frag(B0 + (size_t)48 * K + k, K);
    acc[0][0] = WMMA_BF16(a0, b0, acc[0][0]);
    acc[0][1] = WMMA_BF16(a0, b1, acc[0][1]);
    acc[0][2] = WMMA_BF16(a0, b2, acc[0][2]);
    acc[0][3] = WMMA_BF16(a0, b3, acc[0][3]);
    acc[1][0] = WMMA_BF16(a1, b0, acc[1][0]);
    acc[1][1] = WMMA_BF16(a1, b1, acc[1][1]);
    acc[1][2] = WMMA_BF16(a1, b2, acc[1][2]);
    acc[1][3] = WMMA_BF16(a1, b3, acc[1][3]);
    acc[2][0] = WMMA_BF16(a2, b0, acc[2][0]);
    acc[2][1] = WMMA_BF16(a2, b1, acc[2][1]);
    acc[2][2] = WMMA_BF16(a2, b2, acc[2][2]);
    acc[2][3] = WMMA_BF16(a2, b3, acc[2][3]);
    acc[3][0] = WMMA_BF16(a3, b0, acc[3][0]);
    acc[3][1] = WMMA_BF16(a3, b1, acc[3][1]);
    acc[3][2] = WMMA_BF16(a3, b2, acc[3][2]);
    acc[3][3] = WMMA_BF16(a3, b3, acc[3][3]);
  }
}

// ---------- kernels ----------
__global__ void init_kernel(unsigned* Mord, float* Z) {
  int t = threadIdx.x;
  if (t < B_DIM) { Mord[t] = 0u; Z[t] = 0.0f; }
}

// gate[h] = sigmoid(vp.v1W[h]+v1b[h]) * tanh(vp.v2W[h]+v2b[h]); vp = sigmoid(v')
__global__ __launch_bounds__(256) void gate_kernel(
    const float* __restrict__ v_prime, const float* __restrict__ v1W,
    const float* __restrict__ v1b, const float* __restrict__ v2W,
    const float* __restrict__ v2b, float* __restrict__ gate) {
  const int h = blockIdx.x * 8 + (threadIdx.x >> 5);
  const int lane = threadIdx.x & 31;
  float d1 = 0.0f, d2 = 0.0f;
  for (int k = lane; k < H_DIM; k += 32) {
    float vp = sigmoidf_(v_prime[k]);
    d1 += vp * v1W[(size_t)h * H_DIM + k];
    d2 += vp * v2W[(size_t)h * H_DIM + k];
  }
  for (int o = 16; o > 0; o >>= 1) {
    d1 += __shfl_xor(d1, o, 32);
    d2 += __shfl_xor(d2, o, 32);
  }
  if (lane == 0) gate[h] = sigmoidf_(d1 + v1b[h]) * tanhf(d2 + v2b[h]);
}

__global__ __launch_bounds__(256) void wconv_kernel(const float* __restrict__ W,
                                                    unsigned short* __restrict__ Wb) {
  size_t i = (size_t)blockIdx.x * 256 + threadIdx.x;   // one float4 each
  float4 v = ((const float4*)W)[i];
  ((uint2*)Wb)[i] = make_uint2(pack2bf(v.x, v.y), pack2bf(v.z, v.w));
}

// Fused: both LayerNorms + kr + vr; one block per (b,s) row of H=1024.
__global__ __launch_bounds__(256) void prep_kernel(
    const float* __restrict__ x,
    const float* __restrict__ qn_w, const float* __restrict__ qn_b,
    const float* __restrict__ kvn_w, const float* __restrict__ kvn_b,
    const float* __restrict__ k_prime, const float* __restrict__ gate,
    float* __restrict__ xqn_f32, unsigned short* __restrict__ xqn_bf,
    unsigned short* __restrict__ kr_bf, unsigned short* __restrict__ vr_bf) {
  __shared__ float s1[256], s2[256];
  const size_t row = blockIdx.x;
  const int t = threadIdx.x;
  const float4 v = ((const float4*)(x + row * H_DIM))[t];
  s1[t] = v.x + v.y + v.z + v.w;
  s2[t] = v.x * v.x + v.y * v.y + v.z * v.z + v.w * v.w;
  __syncthreads();
  for (int o = 128; o > 0; o >>= 1) {
    if (t < o) { s1[t] += s1[t + o]; s2[t] += s2[t + o]; }
    __syncthreads();
  }
  const float mean = s1[0] * (1.0f / H_DIM);
  const float var  = s2[0] * (1.0f / H_DIM) - mean * mean;
  const float rs   = rsqrtf(var + EPSV);

  const float4 qw  = ((const float4*)qn_w)[t],  qb4 = ((const float4*)qn_b)[t];
  const float4 kw  = ((const float4*)kvn_w)[t], kb4 = ((const float4*)kvn_b)[t];
  const float4 kp  = ((const float4*)k_prime)[t];
  const float4 g   = ((const float4*)gate)[t];

#define LN_(c, w_, b_) (((c) - mean) * rs * (w_) + (b_))
  float q0 = LN_(v.x, qw.x, qb4.x), q1 = LN_(v.y, qw.y, qb4.y);
  float q2 = LN_(v.z, qw.z, qb4.z), q3 = LN_(v.w, qw.w, qb4.w);
  float n0 = LN_(v.x, kw.x, kb4.x), n1 = LN_(v.y, kw.y, kb4.y);
  float n2 = LN_(v.z, kw.z, kb4.z), n3 = LN_(v.w, kw.w, kb4.w);
#undef LN_
  float k0 = n0 * sigmoidf_(kp.x), k1 = n1 * sigmoidf_(kp.y);
  float k2 = n2 * sigmoidf_(kp.z), k3 = n3 * sigmoidf_(kp.w);
  float r0 = n0 * g.x, r1 = n1 * g.y, r2 = n2 * g.z, r3 = n3 * g.w;

  const size_t o4 = row * (H_DIM / 4) + t;
  ((float4*)xqn_f32)[o4] = make_float4(q0, q1, q2, q3);
  ((uint2*)xqn_bf)[o4]   = make_uint2(pack2bf(q0, q1), pack2bf(q2, q3));
  ((uint2*)kr_bf)[o4]    = make_uint2(pack2bf(k0, k1), pack2bf(k2, k3));
  ((uint2*)vr_bf)[o4]    = make_uint2(pack2bf(r0, r1), pack2bf(r2, r3));
}

// vr[b][s][h] -> vrT[b][h][s]  (LDS 32x32 tile, +1 pad against bank conflicts)
__global__ __launch_bounds__(256) void transpose_kernel(
    const unsigned short* __restrict__ vr, unsigned short* __restrict__ vrT) {
  __shared__ unsigned short tile[32][33];
  const int b = blockIdx.z;
  const int s0 = blockIdx.x * 32, h0 = blockIdx.y * 32;
  const int tx = threadIdx.x & 31, ty = threadIdx.x >> 5;
  const unsigned short* src = vr + (size_t)b * S_DIM * H_DIM;
  unsigned short*       dst = vrT + (size_t)b * H_DIM * S_DIM;
#pragma unroll
  for (int j = 0; j < 4; j++) {
    int r = ty + j * 8;
    tile[r][tx] = src[(size_t)(s0 + r) * H_DIM + h0 + tx];
  }
  __syncthreads();
#pragma unroll
  for (int j = 0; j < 4; j++) {
    int r = ty + j * 8;
    dst[(size_t)(h0 + r) * S_DIM + s0 + tx] = tile[tx][r];
  }
}

// qr = (xqn @ qW^T + qb) * sigmoid(q') ; output bf16 [M=B*S, N=H]
// Block tile 256x128: 8 waves as 4(M) x 2(N), each wave 64x64.
__global__ __launch_bounds__(256) void gemm_qr_kernel(
    const unsigned short* __restrict__ A, const unsigned short* __restrict__ Bm,
    const float* __restrict__ qb, const float* __restrict__ q_prime,
    unsigned short* __restrict__ Q) {
  const int K = H_DIM, N = H_DIM;
  const int wid = threadIdx.x >> 5, wm = wid & 3, wn = wid >> 2;
  const int m0 = blockIdx.x * 256 + wm * 64;
  const int n0 = blockIdx.y * 128 + wn * 64;
  v8f acc[4][4] = {};
  gemm_tile_loop(A + (size_t)m0 * K, Bm + (size_t)n0 * K, K, acc);
  const int lane = threadIdx.x & 31, r = lane & 15, half = lane >> 4;
#pragma unroll
  for (int i = 0; i < 4; i++)
#pragma unroll
    for (int t = 0; t < 4; t++) {
      const int col = n0 + 16 * t + r;
      const float bias = qb[col], s = sigmoidf_(q_prime[col]);
#pragma unroll
      for (int j = 0; j < 8; j++) {
        const int row = m0 + 16 * i + j + 8 * half;
        Q[(size_t)row * N + col] = f2bf((acc[i][t][j] + bias) * s);
      }
    }
}

// scores[b] = qr[b] @ kr[b]^T / sqrt(H), fp32; + per-batch max via ordered-uint atomicMax
__global__ __launch_bounds__(256) void gemm_scores_kernel(
    const unsigned short* __restrict__ Q, const unsigned short* __restrict__ Kr,
    float* __restrict__ scores, unsigned* __restrict__ Mord) {
  __shared__ float red[256];
  const int K = H_DIM;
  const int b = blockIdx.z;
  const int wid = threadIdx.x >> 5, wm = wid & 3, wn = wid >> 2;
  const int m0 = blockIdx.x * 256 + wm * 64;
  const int n0 = blockIdx.y * 128 + wn * 64;
  v8f acc[4][4] = {};
  gemm_tile_loop(Q + (size_t)b * S_DIM * H_DIM + (size_t)m0 * K,
                 Kr + (size_t)b * S_DIM * H_DIM + (size_t)n0 * K, K, acc);
  float* C = scores + (size_t)b * S_DIM * S_DIM;
  const float scale = 0.03125f;  // 1/sqrt(1024)
  const int lane = threadIdx.x & 31, r = lane & 15, half = lane >> 4;
  float mx = -3.0e38f;
#pragma unroll
  for (int i = 0; i < 4; i++)
#pragma unroll
    for (int t = 0; t < 4; t++)
#pragma unroll
      for (int j = 0; j < 8; j++) {
        const int row = m0 + 16 * i + j + 8 * half;
        const int col = n0 + 16 * t + r;
        const float val = acc[i][t][j] * scale;
        C[(size_t)row * S_DIM + col] = val;
        mx = fmaxf(mx, val);
      }
  red[threadIdx.x] = mx;
  __syncthreads();
  for (int o = 128; o > 0; o >>= 1) {
    if (threadIdx.x < o) red[threadIdx.x] = fmaxf(red[threadIdx.x], red[threadIdx.x + o]);
    __syncthreads();
  }
  if (threadIdx.x == 0) atomicMax(&Mord[b], f32_ord(red[0]));
}

// P = exp(scores - M_b) as bf16; block partial sums (deterministic Z reduction)
__global__ __launch_bounds__(256) void expsum_kernel(
    const float* __restrict__ scores, const unsigned* __restrict__ Mord,
    unsigned short* __restrict__ P, float* __restrict__ partials) {
  __shared__ float red[256];
  const int b = blockIdx.y;
  const float M = ord_f32(Mord[b]);
  const size_t base = (size_t)b * S_DIM * S_DIM;
  const size_t idx = (size_t)blockIdx.x * 1024 + (size_t)threadIdx.x * 4;
  const float4 v = *(const float4*)(scores + base + idx);
  const float e0 = __expf(v.x - M), e1 = __expf(v.y - M);
  const float e2 = __expf(v.z - M), e3 = __expf(v.w - M);
  ((uint2*)P)[(base + idx) >> 2] = make_uint2(pack2bf(e0, e1), pack2bf(e2, e3));
  red[threadIdx.x] = e0 + e1 + e2 + e3;
  __syncthreads();
  for (int o = 128; o > 0; o >>= 1) {
    if (threadIdx.x < o) red[threadIdx.x] += red[threadIdx.x + o];
    __syncthreads();
  }
  if (threadIdx.x == 0) partials[(size_t)b * 4096 + blockIdx.x] = red[0];
}

__global__ __launch_bounds__(256) void zreduce_kernel(const float* __restrict__ partials,
                                                      float* __restrict__ Z) {
  __shared__ float red[256];
  const int b = blockIdx.x;
  float s = 0.0f;
  for (int i = threadIdx.x; i < 4096; i += 256) s += partials[(size_t)b * 4096 + i];
  red[threadIdx.x] = s;
  __syncthreads();
  for (int o = 128; o > 0; o >>= 1) {
    if (threadIdx.x < o) red[threadIdx.x] += red[threadIdx.x + o];
    __syncthreads();
  }
  if (threadIdx.x == 0) Z[b] = red[0];
}

// out = (P @ vr) / Z_b + xqn   (B of the GEMM is vrT[b] stored [N=h, K=t])
__global__ __launch_bounds__(256) void gemm_attn_kernel(
    const unsigned short* __restrict__ P, const unsigned short* __restrict__ VT,
    const float* __restrict__ Z, const float* __restrict__ xqn,
    float* __restrict__ out) {
  const int K = S_DIM;
  const int b = blockIdx.z;
  const int wid = threadIdx.x >> 5, wm = wid & 3, wn = wid >> 2;
  const int m0 = blockIdx.x * 256 + wm * 64;
  const int n0 = blockIdx.y * 128 + wn * 64;
  v8f acc[4][4] = {};
  gemm_tile_loop(P + (size_t)b * S_DIM * S_DIM + (size_t)m0 * K,
                 VT + (size_t)b * H_DIM * S_DIM + (size_t)n0 * K, K, acc);
  const float invZ = 1.0f / Z[b];
  const int lane = threadIdx.x & 31, r = lane & 15, half = lane >> 4;
#pragma unroll
  for (int i = 0; i < 4; i++)
#pragma unroll
    for (int t = 0; t < 4; t++)
#pragma unroll
      for (int j = 0; j < 8; j++) {
        const int row = m0 + 16 * i + j + 8 * half;
        const int col = n0 + 16 * t + r;
        const size_t gi = ((size_t)b * S_DIM + row) * H_DIM + col;
        out[gi] = acc[i][t][j] * invZ + xqn[gi];
      }
}

// ---------- launch ----------
extern "C" void kernel_launch(void* const* d_in, const int* in_sizes, int n_in,
                              void* d_out, int out_size, void* d_ws, size_t ws_size,
                              hipStream_t stream) {
  (void)in_sizes; (void)n_in; (void)out_size; (void)ws_size;
  const float* x       = (const float*)d_in[0];
  const float* qn_w    = (const float*)d_in[1];
  const float* qn_b    = (const float*)d_in[2];
  const float* kvn_w   = (const float*)d_in[3];
  const float* kvn_b   = (const float*)d_in[4];
  const float* q_prime = (const float*)d_in[5];
  const float* k_prime = (const float*)d_in[6];
  const float* v_prime = (const float*)d_in[7];
  const float* qW      = (const float*)d_in[8];
  const float* qb      = (const float*)d_in[9];
  const float* v1W     = (const float*)d_in[10];
  const float* v1b     = (const float*)d_in[11];
  const float* v2W     = (const float*)d_in[12];
  const float* v2b     = (const float*)d_in[13];
  float* out = (float*)d_out;

  char* w = (char*)d_ws;
  size_t off = 0;
  auto take = [&](size_t bytes) {
    char* p = w + off;
    off += (bytes + 255) & ~(size_t)255;
    return p;
  };
  const size_t MBS = (size_t)B_DIM * S_DIM;                    // 16384 rows
  float*          xqn_f32 = (float*)take(MBS * H_DIM * 4);     // 67 MB (residual)
  unsigned short* kr_bf   = (unsigned short*)take(MBS * H_DIM * 2);
  unsigned short* vrT_bf  = (unsigned short*)take(MBS * H_DIM * 2);
  unsigned short* qr_bf   = (unsigned short*)take(MBS * H_DIM * 2);
  float*          scores  = (float*)take((size_t)B_DIM * S_DIM * S_DIM * 4);  // 134 MB, L2-resident
  unsigned short* P_bf    = (unsigned short*)take((size_t)B_DIM * S_DIM * S_DIM * 2);
  unsigned short* qWb     = (unsigned short*)take((size_t)H_DIM * H_DIM * 2);
  float*          gate    = (float*)take(H_DIM * 4);
  float*          parts   = (float*)take((size_t)B_DIM * 4096 * 4);
  unsigned*       Mord    = (unsigned*)take(256);
  float*          Zf      = (float*)take(256);
  // Aliases inside the P region (both dead before expsum writes P):
  unsigned short* vr_bf  = P_bf;                  // consumed by transpose
  unsigned short* xqn_bf = P_bf + MBS * H_DIM;    // consumed by gemm_qr

  init_kernel<<<1, 32, 0, stream>>>(Mord, Zf);
  gate_kernel<<<H_DIM / 8, 256, 0, stream>>>(v_prime, v1W, v1b, v2W, v2b, gate);
  wconv_kernel<<<(H_DIM * H_DIM) / 1024, 256, 0, stream>>>(qW, qWb);
  prep_kernel<<<(unsigned)MBS, 256, 0, stream>>>(x, qn_w, qn_b, kvn_w, kvn_b,
                                                 k_prime, gate, xqn_f32, xqn_bf,
                                                 kr_bf, vr_bf);
  transpose_kernel<<<dim3(S_DIM / 32, H_DIM / 32, B_DIM), 256, 0, stream>>>(vr_bf, vrT_bf);
  gemm_qr_kernel<<<dim3(MBS / 256, H_DIM / 128), 256, 0, stream>>>(xqn_bf, qWb, qb,
                                                                   q_prime, qr_bf);
  gemm_scores_kernel<<<dim3(S_DIM / 256, S_DIM / 128, B_DIM), 256, 0, stream>>>(
      qr_bf, kr_bf, scores, Mord);
  expsum_kernel<<<dim3((S_DIM * S_DIM) / 1024, B_DIM), 256, 0, stream>>>(scores, Mord,
                                                                         P_bf, parts);
  zreduce_kernel<<<B_DIM, 256, 0, stream>>>(parts, Zf);
  gemm_attn_kernel<<<dim3(S_DIM / 256, H_DIM / 128, B_DIM), 256, 0, stream>>>(
      P_bf, vrT_bf, Zf, xqn_f32, out);
}
